// PrunableQwenImageTransformerBlock_47691316855532
// MI455X (gfx1250) — matrix operand order repaired and checked by reading, not compiled, not run on hardware
//
#include <hip/hip_runtime.h>
#include <hip/hip_bf16.h>

// ---------------------------------------------------------------------------
// Types / helpers
// ---------------------------------------------------------------------------
typedef __bf16 bf16;
typedef __attribute__((ext_vector_type(16))) __bf16 v16bf;
typedef __attribute__((ext_vector_type(8)))  __bf16 v8bf;
typedef __attribute__((ext_vector_type(4)))  __bf16 v4bf;
typedef __attribute__((ext_vector_type(8)))  float  v8f;
typedef int v4i __attribute__((vector_size(16)));

// Optional CDNA5 async global->LDS staging (ASYNCcnt-tracked), with fallback.
#if defined(__has_builtin)
# if __has_builtin(__builtin_amdgcn_global_load_async_to_lds_b128) && \
     __has_builtin(__builtin_amdgcn_s_wait_asynccnt)
#  define HAVE_ASYNC_LDS 1
# endif
#endif
#ifndef HAVE_ASYNC_LDS
# define HAVE_ASYNC_LDS 0
#endif

#if HAVE_ASYNC_LDS
#define ASYNC_B128(gp, lp)                                                  \
  __builtin_amdgcn_global_load_async_to_lds_b128(                           \
      (__attribute__((address_space(1))) v4i*)(gp),                         \
      (__attribute__((address_space(3))) v4i*)(lp), 0, 0)
#define ASYNC_WAIT() __builtin_amdgcn_s_wait_asynccnt(0)
#endif

__device__ __forceinline__ bf16 f2bf(float f) {
  unsigned u = __builtin_bit_cast(unsigned, f);
  u += 0x7FFFu + ((u >> 16) & 1u);            // round-to-nearest-even
  unsigned short h = (unsigned short)(u >> 16);
  return __builtin_bit_cast(bf16, h);
}

__device__ __forceinline__ v16bf frag16(const bf16* lo, const bf16* hi) {
  v8bf a = *(const v8bf*)lo;
  v8bf b = *(const v8bf*)hi;
  return __builtin_shufflevector(a, b, 0,1,2,3,4,5,6,7,8,9,10,11,12,13,14,15);
}

__device__ __forceinline__ v8f wmma_bf16(v16bf a, v16bf b, v8f c) {
  return __builtin_amdgcn_wmma_f32_16x16x32_bf16(false, a, false, b, (short)0, c,
                                                 false, false);
}

// Problem constants
#define DD   3072
#define LL   3072
#define LD   768
#define TT   256
#define HH   24
#define HDM  128
#define DFF  12288
#define LQ   (TT + LD)   // 1024
#define LK   (TT + LL)   // 3328

// ---------------------------------------------------------------------------
// silu(temb)
// ---------------------------------------------------------------------------
__global__ void k_silu(const float* __restrict__ temb, float* __restrict__ s) {
  int i = blockIdx.x * 256 + threadIdx.x;
  if (i < DD) { float v = temb[i]; s[i] = v / (1.0f + __expf(-v)); }
}

// ---------------------------------------------------------------------------
// fp32 -> bf16 weight pre-conversion (x4 vectorized)
// ---------------------------------------------------------------------------
__global__ void k_f2bf4(const float* __restrict__ src, bf16* __restrict__ dst, int n4) {
  int i = blockIdx.x * 256 + threadIdx.x;
  if (i < n4) {
    float4 v = ((const float4*)src)[i];
    v4bf r;
    r[0] = f2bf(v.x); r[1] = f2bf(v.y); r[2] = f2bf(v.z); r[3] = f2bf(v.w);
    ((v4bf*)dst)[i] = r;
  }
}

// ---------------------------------------------------------------------------
// mod vectors: m = W_mod @ s + b  (one block per output, coalesced reduce)
// ---------------------------------------------------------------------------
__global__ __launch_bounds__(256) void k_modvec(
    const float* __restrict__ Wi, const float* __restrict__ bi,
    const float* __restrict__ Wt, const float* __restrict__ bt,
    const float* __restrict__ s, float* __restrict__ mi, float* __restrict__ mt) {
  int i = blockIdx.x;                 // 0 .. 12*D-1
  const int SIXD = 6 * DD;
  const float* w; const float* b; float* o; int r;
  if (i < SIXD) { r = i;         w = Wi + (size_t)r * DD; b = bi; o = mi; }
  else          { r = i - SIXD;  w = Wt + (size_t)r * DD; b = bt; o = mt; }
  __shared__ float red[256];
  float acc = 0.f;
  for (int k = threadIdx.x; k < DD; k += 256) acc += w[k] * s[k];
  red[threadIdx.x] = acc;
  __syncthreads();
  for (int st = 128; st > 0; st >>= 1) {
    if (threadIdx.x < st) red[threadIdx.x] += red[threadIdx.x + st];
    __syncthreads();
  }
  if (threadIdx.x == 0) o[r] = red[0] + b[r];
}

// ---------------------------------------------------------------------------
// LayerNorm (no affine) + modulate(shift,scale) -> bf16
// ---------------------------------------------------------------------------
__global__ __launch_bounds__(256) void k_lnmod(const float* __restrict__ X,
                                               const float* __restrict__ mod,
                                               bf16* __restrict__ Y) {
  int row = blockIdx.x;
  const float* x = X + (size_t)row * DD;
  __shared__ float s_sum[256];
  __shared__ float s_sq[256];
  float sum = 0.f, sq = 0.f;
  for (int i = threadIdx.x; i < DD; i += 256) { float v = x[i]; sum += v; sq += v * v; }
  s_sum[threadIdx.x] = sum; s_sq[threadIdx.x] = sq;
  __syncthreads();
  for (int st = 128; st > 0; st >>= 1) {
    if (threadIdx.x < st) { s_sum[threadIdx.x] += s_sum[threadIdx.x + st];
                            s_sq[threadIdx.x]  += s_sq[threadIdx.x + st]; }
    __syncthreads();
  }
  float mean = s_sum[0] * (1.0f / DD);
  float var  = s_sq[0] * (1.0f / DD) - mean * mean;
  float inv  = rsqrtf(var + 1e-6f);
  bf16* y = Y + (size_t)row * DD;
  for (int i = threadIdx.x; i < DD; i += 256) {
    float v = (x[i] - mean) * inv;
    v = v * (1.0f + mod[DD + i]) + mod[i];
    y[i] = f2bf(v);
  }
}

// ---------------------------------------------------------------------------
// WMMA GEMM: C[M,N] = A[M,K](bf16) @ W[N,K](bf16)^T + bias
// Block tile 128x128, 8 waves; wave computes 32x64 = 8 WMMA tiles per K-step.
// EPI: 0 = f32,  1 = bf16,  2 = gelu(tanh)->bf16,  3 = res + gate*(.) -> f32
// ---------------------------------------------------------------------------
template <int EPI>
__global__ __launch_bounds__(256) void k_gemm(
    const bf16* __restrict__ A, int lda,
    const bf16* __restrict__ W, int ldw,
    const float* __restrict__ bias,
    float* __restrict__ Cf, bf16* __restrict__ Cb, int ldc,
    const float* __restrict__ res, int ldres,
    const float* __restrict__ gate, int K) {
  __shared__ __align__(16) bf16 lA[128][40];
  __shared__ __align__(16) bf16 lB[128][40];
  const int tid  = threadIdx.x;
  const int lane = tid & 31, wv = tid >> 5;
  const int wm = wv & 3, wn = wv >> 2;               // wave tile: rows wm*32, cols wn*64
  const int m0 = blockIdx.y * 128, n0 = blockIdx.x * 128;
  const int srow = tid >> 1, skq = (tid & 1) * 16;   // staging: 128 rows x 32 cols

  const int l15 = lane & 15;
  const int kcA = (lane & 16) ? 8 : 0;
  const int kcB = (lane & 16) ? 8 : 0;
  const int ar0 = wm * 32 + l15, ar1 = ar0 + 16;
  const int br0 = wn * 64 + l15, br1 = br0 + 16, br2 = br0 + 32, br3 = br0 + 48;

  v8f c00 = {}, c01 = {}, c02 = {}, c03 = {};
  v8f c10 = {}, c11 = {}, c12 = {}, c13 = {};

  const bf16* gA = &A[(size_t)(m0 + srow) * lda + skq];
  const bf16* gW = &W[(size_t)(n0 + srow) * ldw + skq];

  for (int kk = 0; kk < K; kk += 32) {
#if HAVE_ASYNC_LDS
    ASYNC_B128(&gA[kk],     &lA[srow][skq]);
    ASYNC_B128(&gA[kk + 8], &lA[srow][skq + 8]);
    ASYNC_B128(&gW[kk],     &lB[srow][skq]);
    ASYNC_B128(&gW[kk + 8], &lB[srow][skq + 8]);
#else
    *(v8bf*)&lA[srow][skq]     = *(const v8bf*)&gA[kk];
    *(v8bf*)&lA[srow][skq + 8] = *(const v8bf*)&gA[kk + 8];
    *(v8bf*)&lB[srow][skq]     = *(const v8bf*)&gW[kk];
    *(v8bf*)&lB[srow][skq + 8] = *(const v8bf*)&gW[kk + 8];
#endif
    if (kk + 32 < K) {      // gfx1250 global_prefetch for the next K tile
      __builtin_prefetch((const void*)&gA[kk + 32], 0, 0);
      __builtin_prefetch((const void*)&gW[kk + 32], 0, 0);
    }
#if HAVE_ASYNC_LDS
    ASYNC_WAIT();
#endif
    __syncthreads();
    v16bf a0 = frag16(&lA[ar0][kcA], &lA[ar0][kcA + 16]);
    v16bf a1 = frag16(&lA[ar1][kcA], &lA[ar1][kcA + 16]);
    v16bf b0 = frag16(&lB[br0][kcB], &lB[br0][kcB + 16]);
    v16bf b1 = frag16(&lB[br1][kcB], &lB[br1][kcB + 16]);
    v16bf b2 = frag16(&lB[br2][kcB], &lB[br2][kcB + 16]);
    v16bf b3 = frag16(&lB[br3][kcB], &lB[br3][kcB + 16]);
    c00 = wmma_bf16(a0, b0, c00); c01 = wmma_bf16(a0, b1, c01);
    c02 = wmma_bf16(a0, b2, c02); c03 = wmma_bf16(a0, b3, c03);
    c10 = wmma_bf16(a1, b0, c10); c11 = wmma_bf16(a1, b1, c11);
    c12 = wmma_bf16(a1, b2, c12); c13 = wmma_bf16(a1, b3, c13);
    __syncthreads();
  }

  const int erow = (lane & 16) ? 8 : 0;
#pragma unroll
  for (int i = 0; i < 2; ++i) {
#pragma unroll
    for (int j = 0; j < 4; ++j) {
      v8f c = (i == 0) ? ((j == 0) ? c00 : (j == 1) ? c01 : (j == 2) ? c02 : c03)
                       : ((j == 0) ? c10 : (j == 1) ? c11 : (j == 2) ? c12 : c13);
      int n = n0 + wn * 64 + j * 16 + l15;
      float bv = bias[n];
      float gv = (EPI == 3) ? gate[n] : 0.f;
#pragma unroll
      for (int e = 0; e < 8; ++e) {
        int m = m0 + wm * 32 + i * 16 + erow + e;
        float v = c[e] + bv;
        if constexpr (EPI == 0) {
          Cf[(size_t)m * ldc + n] = v;
        } else if constexpr (EPI == 1) {
          Cb[(size_t)m * ldc + n] = f2bf(v);
        } else if constexpr (EPI == 2) {
          float g = 0.5f * v * (1.0f + tanhf(0.7978845608028654f * (v + 0.044715f * v * v * v)));
          Cb[(size_t)m * ldc + n] = f2bf(g);
        } else {
          Cf[(size_t)m * ldc + n] = res[(size_t)m * ldres + n] + gv * v;
        }
      }
    }
  }
}

// ---------------------------------------------------------------------------
// per-(row, head) RMSNorm + RoPE -> bf16 joint Q/K buffers
// ---------------------------------------------------------------------------
__global__ __launch_bounds__(64) void k_rmsrope(const float* __restrict__ src,
                                                const float* __restrict__ nw,
                                                const float* __restrict__ cosp,
                                                const float* __restrict__ sinp,
                                                bf16* __restrict__ dst) {
  int r = blockIdx.x, h = blockIdx.y, t = threadIdx.x;
  const float* x = src + (size_t)r * DD + h * HDM;
  float x0 = x[2 * t], x1 = x[2 * t + 1];
  __shared__ float red[64];
  red[t] = x0 * x0 + x1 * x1;
  __syncthreads();
  for (int st = 32; st > 0; st >>= 1) {
    if (t < st) red[t] += red[t + st];
    __syncthreads();
  }
  float inv = rsqrtf(red[0] * (1.0f / HDM) + 1e-6f);
  float y0 = x0 * inv * nw[2 * t];
  float y1 = x1 * inv * nw[2 * t + 1];
  float c = cosp[r * 64 + t], s = sinp[r * 64 + t];
  bf16* o = dst + (size_t)r * DD + h * HDM;
  o[2 * t]     = f2bf(y0 * c - y1 * s);
  o[2 * t + 1] = f2bf(y0 * s + y1 * c);
}

// ---------------------------------------------------------------------------
// Flash attention: per (64-q-tile, head). Online softmax, WMMA for QK^T & PV.
// ---------------------------------------------------------------------------
__global__ __launch_bounds__(256) void k_flash(const bf16* __restrict__ JQ,
                                               const bf16* __restrict__ JK,
                                               const bf16* __restrict__ JV,
                                               bf16* __restrict__ O, int Lk) {
  __shared__ __align__(16) bf16 Qt[64][136];
  __shared__ __align__(16) bf16 Kt[64][136];
  __shared__ __align__(16) bf16 VtT[128][72];   // transposed V tile: [hd][kv]
  __shared__ __align__(16) float Sld[64][68];
  __shared__ __align__(16) bf16 Pld[64][72];
  __shared__ float alpha_s[64];
  __shared__ float lsum_s[64];

  const int tid = threadIdx.x, lane = tid & 31, wv = tid >> 5;
  const int h = blockIdx.y, q0 = blockIdx.x * 64;
  const size_t hoff = (size_t)h * HDM;
  const int l15 = lane & 15;
  const int rm = wv & 3, cn = wv >> 2;
  const int kcA = (lane & 16) ? 8 : 0;
  const int kcB = (lane & 16) ? 8 : 0;
  const float scale = 0.08838834764831845f;  // 1/sqrt(128)

  // load Q tile (64 x 128 bf16) — waited inside first loop iteration
#pragma unroll
  for (int i = 0; i < 4; ++i) {
    int c = tid + i * 256;
    int rr = c >> 4, kq = (c & 15) * 8;
#if HAVE_ASYNC_LDS
    ASYNC_B128(&JQ[(size_t)(q0 + rr) * DD + hoff + kq], &Qt[rr][kq]);
#else
    *(v8bf*)&Qt[rr][kq] = *(const v8bf*)&JQ[(size_t)(q0 + rr) * DD + hoff + kq];
#endif
  }

  float m_t = -1e30f, l_t = 0.0f;
  v8f o0 = {}, o1 = {}, o2 = {}, o3 = {};

  for (int kt = 0; kt < Lk; kt += 64) {
    __syncthreads();
    // stage K tile
#pragma unroll
    for (int i = 0; i < 4; ++i) {
      int c = tid + i * 256;
      int rr = c >> 4, kq = (c & 15) * 8;
#if HAVE_ASYNC_LDS
      ASYNC_B128(&JK[(size_t)(kt + rr) * DD + hoff + kq], &Kt[rr][kq]);
#else
      *(v8bf*)&Kt[rr][kq] = *(const v8bf*)&JK[(size_t)(kt + rr) * DD + hoff + kq];
#endif
    }
    // stage V tile transposed
#pragma unroll
    for (int i = 0; i < 32; ++i) {
      int idx = tid + i * 256;
      int rr = idx >> 7, cc = idx & 127;
      VtT[cc][rr] = JV[(size_t)(kt + rr) * DD + hoff + cc];
    }
#if HAVE_ASYNC_LDS
    ASYNC_WAIT();
#endif
    __syncthreads();

    // S = Q @ K^T
    v8f s0 = {}, s1 = {};
#pragma unroll
    for (int ks = 0; ks < 4; ++ks) {
      int kb = ks * 32;
      v16bf a  = frag16(&Qt[rm * 16 + l15][kb + kcA], &Qt[rm * 16 + l15][kb + kcA + 16]);
      v16bf b0 = frag16(&Kt[cn * 32 + l15][kb + kcB], &Kt[cn * 32 + l15][kb + kcB + 16]);
      v16bf b1 = frag16(&Kt[cn * 32 + 16 + l15][kb + kcB], &Kt[cn * 32 + 16 + l15][kb + kcB + 16]);
      s0 = wmma_bf16(a, b0, s0);
      s1 = wmma_bf16(a, b1, s1);
    }
    {
      int mrow = rm * 16 + ((lane & 16) ? 8 : 0);
#pragma unroll
      for (int e = 0; e < 8; ++e) {
        Sld[mrow + e][cn * 32 + l15]      = s0[e] * scale;
        Sld[mrow + e][cn * 32 + 16 + l15] = s1[e] * scale;
      }
    }
    __syncthreads();

    // online softmax (one thread per q-row)
    if (tid < 64) {
      float mx = m_t;
      for (int c2 = 0; c2 < 64; ++c2) mx = fmaxf(mx, Sld[tid][c2]);
      float al = __expf(m_t - mx);
      float sum = 0.f;
      for (int c2 = 0; c2 < 64; ++c2) {
        float p = __expf(Sld[tid][c2] - mx);
        Pld[tid][c2] = f2bf(p);
        sum += p;
      }
      l_t = l_t * al + sum;
      m_t = mx;
      alpha_s[tid] = al;
    }
    __syncthreads();

    // rescale O accumulators and accumulate P @ V
    {
      int rb = rm * 16 + ((lane & 16) ? 8 : 0);
#pragma unroll
      for (int e = 0; e < 8; ++e) {
        float al = alpha_s[rb + e];
        o0[e] *= al; o1[e] *= al; o2[e] *= al; o3[e] *= al;
      }
    }
#pragma unroll
    for (int ks = 0; ks < 2; ++ks) {
      int kb = ks * 32;
      v16bf a = frag16(&Pld[rm * 16 + l15][kb + kcA], &Pld[rm * 16 + l15][kb + kcA + 16]);
      v16bf b0 = frag16(&VtT[cn * 64 + 0 * 16 + l15][kb + kcB], &VtT[cn * 64 + 0 * 16 + l15][kb + kcB + 16]);
      v16bf b1 = frag16(&VtT[cn * 64 + 1 * 16 + l15][kb + kcB], &VtT[cn * 64 + 1 * 16 + l15][kb + kcB + 16]);
      v16bf b2 = frag16(&VtT[cn * 64 + 2 * 16 + l15][kb + kcB], &VtT[cn * 64 + 2 * 16 + l15][kb + kcB + 16]);
      v16bf b3 = frag16(&VtT[cn * 64 + 3 * 16 + l15][kb + kcB], &VtT[cn * 64 + 3 * 16 + l15][kb + kcB + 16]);
      o0 = wmma_bf16(a, b0, o0);
      o1 = wmma_bf16(a, b1, o1);
      o2 = wmma_bf16(a, b2, o2);
      o3 = wmma_bf16(a, b3, o3);
    }
  }

  __syncthreads();
  if (tid < 64) lsum_s[tid] = l_t;
  __syncthreads();

  const int rb = rm * 16 + ((lane & 16) ? 8 : 0);
#pragma unroll
  for (int j = 0; j < 4; ++j) {
    v8f oo = (j == 0) ? o0 : (j == 1) ? o1 : (j == 2) ? o2 : o3;
    int col = cn * 64 + j * 16 + l15;
#pragma unroll
    for (int e = 0; e < 8; ++e) {
      int row = rb + e;
      O[(size_t)(q0 + row) * DD + hoff + col] = f2bf(oo[e] / lsum_s[row]);
    }
  }
}

// ---------------------------------------------------------------------------
// tail copy
// ---------------------------------------------------------------------------
__global__ void k_copy(const float* __restrict__ src, float* __restrict__ dst, int n) {
  int i = blockIdx.x * 256 + threadIdx.x;
  if (i < n) dst[i] = src[i];
}

// ---------------------------------------------------------------------------
// Orchestration
// ---------------------------------------------------------------------------
extern "C" void kernel_launch(void* const* d_in, const int* in_sizes, int n_in,
                              void* d_out, int out_size, void* d_ws, size_t ws_size,
                              hipStream_t stream) {
  (void)in_sizes; (void)n_in; (void)out_size; (void)ws_size;
  const float* hs       = (const float*)d_in[0];
  const float* ehs      = (const float*)d_in[1];
  const float* temb     = (const float*)d_in[2];
  const float* img_cos  = (const float*)d_in[3];
  const float* img_sin  = (const float*)d_in[4];
  const float* txt_cos  = (const float*)d_in[5];
  const float* txt_sin  = (const float*)d_in[6];
  const float* w_img_mod = (const float*)d_in[7];
  const float* b_img_mod = (const float*)d_in[8];
  const float* w_txt_mod = (const float*)d_in[9];
  const float* b_txt_mod = (const float*)d_in[10];
  const float* w_q = (const float*)d_in[11];      const float* b_q = (const float*)d_in[12];
  const float* w_add_q = (const float*)d_in[13];  const float* b_add_q = (const float*)d_in[14];
  const float* w_k = (const float*)d_in[15];      const float* b_k = (const float*)d_in[16];
  const float* w_add_k = (const float*)d_in[17];  const float* b_add_k = (const float*)d_in[18];
  const float* w_v = (const float*)d_in[19];      const float* b_v = (const float*)d_in[20];
  const float* w_add_v = (const float*)d_in[21];  const float* b_add_v = (const float*)d_in[22];
  const float* norm_q_w = (const float*)d_in[23];
  const float* norm_k_w = (const float*)d_in[24];
  const float* norm_added_q_w = (const float*)d_in[25];
  const float* norm_added_k_w = (const float*)d_in[26];
  const float* w_out = (const float*)d_in[27];    const float* b_out = (const float*)d_in[28];
  const float* w_add_out = (const float*)d_in[29];const float* b_add_out = (const float*)d_in[30];
  const float* w_mlp1 = (const float*)d_in[31];   const float* b_mlp1 = (const float*)d_in[32];
  const float* w_mlp2 = (const float*)d_in[33];   const float* b_mlp2 = (const float*)d_in[34];
  const float* w_tmlp1 = (const float*)d_in[35];  const float* b_tmlp1 = (const float*)d_in[36];
  const float* w_tmlp2 = (const float*)d_in[37];  const float* b_tmlp2 = (const float*)d_in[38];

  float* out = (float*)d_out;
  char* ws = (char*)d_ws;
  size_t off = 0;
  auto alloc = [&](size_t bytes) -> void* {
    void* p = ws + off;
    off = (off + bytes + 255) & ~(size_t)255;
    return p;
  };
  float* sv      = (float*)alloc((size_t)DD * 4);
  float* mod_img = (float*)alloc((size_t)6 * DD * 4);
  float* mod_txt = (float*)alloc((size_t)6 * DD * 4);
  bf16* img_in   = (bf16*)alloc((size_t)LL * DD * 2);
  bf16* txt_in   = (bf16*)alloc((size_t)TT * DD * 2);
  float* qf      = (float*)alloc((size_t)LD * DD * 4);
  float* kf      = (float*)alloc((size_t)LL * DD * 4);
  float* tqf     = (float*)alloc((size_t)TT * DD * 4);
  float* tkf     = (float*)alloc((size_t)TT * DD * 4);
  bf16* jq       = (bf16*)alloc((size_t)LQ * DD * 2);
  bf16* jk       = (bf16*)alloc((size_t)LK * DD * 2);
  bf16* jv       = (bf16*)alloc((size_t)LK * DD * 2);
  bf16* attn     = (bf16*)alloc((size_t)LQ * DD * 2);
  float* den     = (float*)alloc((size_t)LD * DD * 4);
  float* txt2    = (float*)alloc((size_t)TT * DD * 4);
  bf16* den_in2  = (bf16*)alloc((size_t)LD * DD * 2);
  bf16* txt_in2  = (bf16*)alloc((size_t)TT * DD * 2);
  bf16* mlph_i   = (bf16*)alloc((size_t)LD * DFF * 2);
  bf16* mlph_t   = (bf16*)alloc((size_t)TT * DFF * 2);
  // bf16 copies of GEMM weights
  const size_t WDD = (size_t)DD * DD, WFF = (size_t)DD * DFF;
  bf16* wq_b   = (bf16*)alloc(WDD * 2);
  bf16* wk_b   = (bf16*)alloc(WDD * 2);
  bf16* wv_b   = (bf16*)alloc(WDD * 2);
  bf16* waq_b  = (bf16*)alloc(WDD * 2);
  bf16* wak_b  = (bf16*)alloc(WDD * 2);
  bf16* wav_b  = (bf16*)alloc(WDD * 2);
  bf16* wo_b   = (bf16*)alloc(WDD * 2);
  bf16* wao_b  = (bf16*)alloc(WDD * 2);
  bf16* wm1_b  = (bf16*)alloc(WFF * 2);
  bf16* wm2_b  = (bf16*)alloc(WFF * 2);
  bf16* wtm1_b = (bf16*)alloc(WFF * 2);
  bf16* wtm2_b = (bf16*)alloc(WFF * 2);

  // 0) one-time weight conversion to bf16 (stream pass; bf16 copies live in L2)
  {
    const int nd4 = (int)(WDD / 4), nf4 = (int)(WFF / 4);
    const int gd = (nd4 + 255) / 256, gf = (nf4 + 255) / 256;
    k_f2bf4<<<gd, 256, 0, stream>>>(w_q, wq_b, nd4);
    k_f2bf4<<<gd, 256, 0, stream>>>(w_k, wk_b, nd4);
    k_f2bf4<<<gd, 256, 0, stream>>>(w_v, wv_b, nd4);
    k_f2bf4<<<gd, 256, 0, stream>>>(w_add_q, waq_b, nd4);
    k_f2bf4<<<gd, 256, 0, stream>>>(w_add_k, wak_b, nd4);
    k_f2bf4<<<gd, 256, 0, stream>>>(w_add_v, wav_b, nd4);
    k_f2bf4<<<gd, 256, 0, stream>>>(w_out, wo_b, nd4);
    k_f2bf4<<<gd, 256, 0, stream>>>(w_add_out, wao_b, nd4);
    k_f2bf4<<<gf, 256, 0, stream>>>(w_mlp1, wm1_b, nf4);
    k_f2bf4<<<gf, 256, 0, stream>>>(w_mlp2, wm2_b, nf4);
    k_f2bf4<<<gf, 256, 0, stream>>>(w_tmlp1, wtm1_b, nf4);
    k_f2bf4<<<gf, 256, 0, stream>>>(w_tmlp2, wtm2_b, nf4);
  }

  // 1) modulation path
  k_silu<<<(DD + 255) / 256, 256, 0, stream>>>(temb, sv);
  k_modvec<<<12 * DD, 256, 0, stream>>>(w_img_mod, b_img_mod, w_txt_mod,
                                        b_txt_mod, sv, mod_img, mod_txt);
  // 2) LN + modulate -> bf16 activations
  k_lnmod<<<LL, 256, 0, stream>>>(hs, mod_img, img_in);
  k_lnmod<<<TT, 256, 0, stream>>>(ehs, mod_txt, txt_in);

  // 3) QKV projections (V goes straight to joint-V as bf16)
  k_gemm<0><<<dim3(DD / 128, LD / 128), 256, 0, stream>>>(img_in, DD, wq_b, DD, b_q, qf, nullptr, DD, nullptr, 0, nullptr, DD);
  k_gemm<0><<<dim3(DD / 128, LL / 128), 256, 0, stream>>>(img_in, DD, wk_b, DD, b_k, kf, nullptr, DD, nullptr, 0, nullptr, DD);
  k_gemm<1><<<dim3(DD / 128, LL / 128), 256, 0, stream>>>(img_in, DD, wv_b, DD, b_v, nullptr, jv + (size_t)TT * DD, DD, nullptr, 0, nullptr, DD);
  k_gemm<0><<<dim3(DD / 128, TT / 128), 256, 0, stream>>>(txt_in, DD, waq_b, DD, b_add_q, tqf, nullptr, DD, nullptr, 0, nullptr, DD);
  k_gemm<0><<<dim3(DD / 128, TT / 128), 256, 0, stream>>>(txt_in, DD, wak_b, DD, b_add_k, tkf, nullptr, DD, nullptr, 0, nullptr, DD);
  k_gemm<1><<<dim3(DD / 128, TT / 128), 256, 0, stream>>>(txt_in, DD, wav_b, DD, b_add_v, nullptr, jv, DD, nullptr, 0, nullptr, DD);

  // 4) RMSNorm + RoPE -> joint bf16 Q/K
  k_rmsrope<<<dim3(TT, HH), 64, 0, stream>>>(tqf, norm_added_q_w, txt_cos, txt_sin, jq);
  k_rmsrope<<<dim3(TT, HH), 64, 0, stream>>>(tkf, norm_added_k_w, txt_cos, txt_sin, jk);
  k_rmsrope<<<dim3(LD, HH), 64, 0, stream>>>(qf, norm_q_w, img_cos, img_sin, jq + (size_t)TT * DD);
  k_rmsrope<<<dim3(LL, HH), 64, 0, stream>>>(kf, norm_k_w, img_cos, img_sin, jk + (size_t)TT * DD);

  // 5) flash attention
  k_flash<<<dim3(LQ / 64, HH), 256, 0, stream>>>(jq, jk, jv, attn, LK);

  // 6) output projections fused with gated residual
  k_gemm<3><<<dim3(DD / 128, TT / 128), 256, 0, stream>>>(attn, DD, wao_b, DD, b_add_out, txt2, nullptr, DD, ehs, DD, mod_txt + 2 * DD, DD);
  k_gemm<3><<<dim3(DD / 128, LD / 128), 256, 0, stream>>>(attn + (size_t)TT * DD, DD, wo_b, DD, b_out, den, nullptr, DD, hs, DD, mod_img + 2 * DD, DD);

  // 7) second LN + modulate
  k_lnmod<<<LD, 256, 0, stream>>>(den, mod_img + 3 * DD, den_in2);
  k_lnmod<<<TT, 256, 0, stream>>>(txt2, mod_txt + 3 * DD, txt_in2);

  // 8) MLPs
  k_gemm<2><<<dim3(DFF / 128, LD / 128), 256, 0, stream>>>(den_in2, DD, wm1_b, DD, b_mlp1, nullptr, mlph_i, DFF, nullptr, 0, nullptr, DD);
  k_gemm<3><<<dim3(DD / 128, LD / 128), 256, 0, stream>>>(mlph_i, DFF, wm2_b, DFF, b_mlp2, out + (size_t)TT * DD, nullptr, DD, den, DD, mod_img + 5 * DD, DFF);
  k_gemm<2><<<dim3(DFF / 128, TT / 128), 256, 0, stream>>>(txt_in2, DD, wtm1_b, DD, b_tmlp1, nullptr, mlph_t, DFF, nullptr, 0, nullptr, DD);
  k_gemm<3><<<dim3(DD / 128, TT / 128), 256, 0, stream>>>(mlph_t, DFF, wtm2_b, DFF, b_tmlp2, out, nullptr, DD, txt2, DD, mod_txt + 5 * DD, DFF);

  // 9) untouched image rows Ld..L
  const int ntail = (LL - LD) * DD;
  k_copy<<<(ntail + 255) / 256, 256, 0, stream>>>(hs + (size_t)LD * DD, out + (size_t)TT * DD + (size_t)LD * DD, ntail);
}
